// ParallelAttention_30906584662633
// MI455X (gfx1250) — matrix operand order repaired and checked by reading, not compiled, not run on hardware
//
#include <hip/hip_runtime.h>

// ---------------- problem constants ----------------
#define S_  2048
#define B_  2
#define H_  2048
#define NP_ 16
#define HN_ 128
#define NT_ (S_ * B_)   // 4096 tokens
#define H3_ (3 * H_)    // 6144

// ---------------- WMMA types / helpers ----------------
typedef __attribute__((ext_vector_type(16))) __bf16 v16bf;
typedef __attribute__((ext_vector_type(8)))  float  v8f;

__device__ __forceinline__ v8f wmma_bf16f32(v16bf a, v16bf b, v8f c) {
  // (neg_a, A, neg_b, B, c_mod, C, reuse_a, reuse_b)
  return __builtin_amdgcn_wmma_f32_16x16x32_bf16(false, a, false, b, (short)0, c,
                                                 false, false);
}

union FragU { v16bf v; uint4 q[2]; };

// A-matrix 16x32 bf16 fragment (ISA 7.12.2): lane l holds M=l%16,
// K = 8*(l/16)+{0..7} and 16+8*(l/16)+{0..7}  -> two 16B chunks.
__device__ __forceinline__ v16bf load_frag_a(const __bf16* base, int row0, int k0, int ld) {
  int lane = threadIdx.x & 31;
  const __bf16* p = base + (long)(row0 + (lane & 15)) * ld + k0 + 8 * (lane >> 4);
  FragU u;
  u.q[0] = *(const uint4*)(p);
  u.q[1] = *(const uint4*)(p + 16);
  return u.v;
}

// B-matrix 32x16 bf16 fragment: lane l holds N=l%16,
// K = 8*(l/16)+{0..7} and 16+8*(l/16)+{0..7}  -> two 16B chunks.
__device__ __forceinline__ v16bf load_frag_b(const __bf16* base, int n0, int k0, int ld) {
  int lane = threadIdx.x & 31;
  const __bf16* p = base + (long)(n0 + (lane & 15)) * ld + k0 + 8 * (lane >> 4);
  FragU u;
  u.q[0] = *(const uint4*)(p);
  u.q[1] = *(const uint4*)(p + 16);
  return u.v;
}

// Async global -> LDS copy, 16B per lane (ASYNCcnt-tracked, ISA 08 §4).
// lds_addr = wave-relative LDS byte offset (low 32 bits of generic pointer).
__device__ __forceinline__ void async_copy_b128(unsigned lds_addr, const void* gaddr) {
  asm volatile("global_load_async_to_lds_b128 %0, %1, off"
               :: "v"(lds_addr), "v"((unsigned long long)(uintptr_t)gaddr)
               : "memory");
}
__device__ __forceinline__ void wait_async_le8() {
  asm volatile("s_wait_asynccnt 0x8" ::: "memory");
}
__device__ __forceinline__ void wait_async_0() {
  asm volatile("s_wait_asynccnt 0x0" ::: "memory");
}

// ---------------- kernels ----------------

__global__ __launch_bounds__(256) void cvt_f32_bf16(const float* __restrict__ src,
                                                    __bf16* __restrict__ dst, int n) {
  int i = blockIdx.x * 256 + threadIdx.x;
  if (i < n) dst[i] = (__bf16)src[i];
}

// C[M,N] = A[M,K] * Bw[N,K]^T + bias[N]   (NT gemm, bf16 in, f32 acc)
// block = 256 threads = 8 waves (4M x 2N); wave tile 32x64; WG tile 128x128.
// K staged in double-buffered LDS tiles of 64 via global_load_async_to_lds_b128;
// next stage's 8 async ops per wave stay in flight while current stage computes.
#define KTILE 64
#define LDSLD 72   // 64 + 8 element pad -> 144B row stride, conflict-free b128 frags

template <bool F32OUT>
__global__ __launch_bounds__(256) void gemm_nt(const __bf16* __restrict__ A,
                                               const __bf16* __restrict__ Bw,
                                               const float* __restrict__ bias,
                                               float* __restrict__ Cf,
                                               __bf16* __restrict__ Cb,
                                               int N, int K) {
  __shared__ __align__(16) __bf16 As[2][128 * LDSLD];
  __shared__ __align__(16) __bf16 Bs[2][128 * LDSLD];

  const int tid = threadIdx.x;
  const int wid = tid >> 5;
  const int wm = wid >> 1, wn = wid & 1;
  const int lane = tid & 31;
  const long arow0 = (long)blockIdx.y * 128;
  const long bcol0 = (long)blockIdx.x * 128;

  // Stage ks..ks+63 of both tiles into buffer `buf`:
  // 1024 chunks of 16B per tile; 256 threads x 4 chunks each (8 async ops/thread).
  auto issue_stage = [&](int buf, int ks) {
#pragma unroll
    for (int i = 0; i < 4; ++i) {
      int c = tid + 256 * i;
      int r = c >> 3, ch = (c & 7) * 8;
      async_copy_b128((unsigned)(uintptr_t)&As[buf][r * LDSLD + ch],
                      A + (arow0 + r) * K + ks + ch);
    }
#pragma unroll
    for (int i = 0; i < 4; ++i) {
      int c = tid + 256 * i;
      int r = c >> 3, ch = (c & 7) * 8;
      async_copy_b128((unsigned)(uintptr_t)&Bs[buf][r * LDSLD + ch],
                      Bw + (bcol0 + r) * K + ks + ch);
    }
  };

  v8f acc[2][4] = {};
  const int nst = K / KTILE;

  issue_stage(0, 0);
  for (int s = 0; s < nst; ++s) {
    const int buf = s & 1;
    if (s + 1 < nst) {
      issue_stage(buf ^ 1, (s + 1) * KTILE);
      wait_async_le8();            // stage s's 8 ops done; stage s+1 in flight
    } else {
      wait_async_0();
    }
    __syncthreads();               // stage-s tile visible to all waves

    const __bf16* as = &As[buf][0];
    const __bf16* bs = &Bs[buf][0];
#pragma unroll
    for (int kk = 0; kk < 2; ++kk) {
      v16bf a0 = load_frag_a(as, wm * 32,      kk * 32, LDSLD);
      v16bf a1 = load_frag_a(as, wm * 32 + 16, kk * 32, LDSLD);
#pragma unroll
      for (int j = 0; j < 4; ++j) {
        v16bf b = load_frag_b(bs, wn * 64 + 16 * j, kk * 32, LDSLD);
        acc[0][j] = wmma_bf16f32(a0, b, acc[0][j]);
        acc[1][j] = wmma_bf16f32(a1, b, acc[1][j]);
      }
    }
    __syncthreads();               // all reads done before buffer is refilled
  }

  const int colx = lane & 15, rb = (lane >> 4) * 8;
  const int m0 = (int)arow0 + wm * 32;
  const int n0 = (int)bcol0 + wn * 64;
#pragma unroll
  for (int i = 0; i < 2; ++i)
#pragma unroll
    for (int j = 0; j < 4; ++j) {
      int col = n0 + 16 * j + colx;
      float bv = bias ? bias[col] : 0.0f;
#pragma unroll
      for (int r = 0; r < 8; ++r) {
        int row = m0 + 16 * i + rb + r;
        float v = acc[i][j][r] + bv;
        if (F32OUT) Cf[(long)row * N + col] = v;
        else        Cb[(long)row * N + col] = (__bf16)v;
      }
    }
}

// mixed[t, 6144] -> per-head Q[b,h,s,d], K[b,h,s,d], V^T[b,h,d,s] (all bf16)
__global__ __launch_bounds__(256) void repack_qkv(const __bf16* __restrict__ mixed,
                                                  __bf16* __restrict__ Qh,
                                                  __bf16* __restrict__ Kh,
                                                  __bf16* __restrict__ VhT) {
  int idx = blockIdx.x * 256 + threadIdx.x;          // (t, h, d)
  if (idx >= NT_ * NP_ * HN_) return;
  int d = idx & (HN_ - 1);
  int h = (idx >> 7) & (NP_ - 1);
  int t = idx >> 11;
  int b = t & (B_ - 1);
  int s = t >> 1;
  long src = (long)t * H3_ + (long)h * (3 * HN_);
  __bf16 q = mixed[src + d];
  __bf16 k = mixed[src + HN_ + d];
  __bf16 v = mixed[src + 2 * HN_ + d];
  long hb = ((long)(b * NP_ + h)) * S_ * HN_;
  Qh[hb + (long)s * HN_ + d] = q;
  Kh[hb + (long)s * HN_ + d] = k;
  VhT[hb + (long)d * S_ + s] = v;
}

// Flash attention: one block per (q-tile of 64 rows, head, batch).
// 4 waves, each owns 16 Q rows; streaming softmax; causal blocks skipped.
__global__ __launch_bounds__(128) void flash_attn(const __bf16* __restrict__ Qh,
                                                  const __bf16* __restrict__ Kh,
                                                  const __bf16* __restrict__ VhT,
                                                  __bf16* __restrict__ Ctx) {
  const int qt  = blockIdx.x;            // 0..S/64-1
  const int h   = blockIdx.y;
  const int b   = blockIdx.z;
  const int wid = threadIdx.x >> 5;
  const int lane = threadIdx.x & 31;
  const int s0 = qt * 64 + wid * 16;

  const long hb = ((long)(b * NP_ + h)) * S_ * HN_;
  const __bf16* q_base = Qh + hb;
  const __bf16* k_base = Kh + hb;
  const __bf16* v_base = VhT + hb;       // [HN_][S_]

  __shared__ __align__(16) __bf16 Pb[4][16][64];  // per-wave P-tile slab (8 KB)

  v16bf qf[4];
#pragma unroll
  for (int kk = 0; kk < 4; ++kk) qf[kk] = load_frag_a(q_base, s0, kk * 32, HN_);

  v8f o[8] = {};
  float mrow[8], lrow[8];
#pragma unroll
  for (int r = 0; r < 8; ++r) { mrow[r] = -1.0e30f; lrow[r] = 0.0f; }

  const float norm = 0.08838834764831845f;  // 1/sqrt(128)
  const int colx = lane & 15, rb = (lane >> 4) * 8;

  for (int kt = 0; kt <= qt; ++kt) {
    const int t0 = kt * 64;

    v8f sf[4] = {};
#pragma unroll
    for (int j = 0; j < 4; ++j)
#pragma unroll
      for (int kk = 0; kk < 4; ++kk)
        sf[j] = wmma_bf16f32(qf[kk], load_frag_b(k_base, t0 + 16 * j, kk * 32, HN_), sf[j]);

    if (kt == qt) {
#pragma unroll
      for (int j = 0; j < 4; ++j)
#pragma unroll
        for (int r = 0; r < 8; ++r) {
          float v = sf[j][r] * norm;
          if (t0 + 16 * j + colx > s0 + rb + r) v = -10000.0f;
          sf[j][r] = v;
        }
    } else {
#pragma unroll
      for (int j = 0; j < 4; ++j)
#pragma unroll
        for (int r = 0; r < 8; ++r) sf[j][r] *= norm;
    }

    float mnew[8], scl[8];
#pragma unroll
    for (int r = 0; r < 8; ++r) {
      float v = fmaxf(fmaxf(sf[0][r], sf[1][r]), fmaxf(sf[2][r], sf[3][r]));
      v = fmaxf(v, __shfl_xor(v, 1, 32));
      v = fmaxf(v, __shfl_xor(v, 2, 32));
      v = fmaxf(v, __shfl_xor(v, 4, 32));
      v = fmaxf(v, __shfl_xor(v, 8, 32));
      mnew[r] = fmaxf(mrow[r], v);
      scl[r]  = __expf(mrow[r] - mnew[r]);
      mrow[r] = mnew[r];
    }
#pragma unroll
    for (int j = 0; j < 4; ++j)
#pragma unroll
      for (int r = 0; r < 8; ++r) sf[j][r] = __expf(sf[j][r] - mnew[r]);
#pragma unroll
    for (int r = 0; r < 8; ++r) {
      float v = sf[0][r] + sf[1][r] + sf[2][r] + sf[3][r];
      v += __shfl_xor(v, 1, 32);
      v += __shfl_xor(v, 2, 32);
      v += __shfl_xor(v, 4, 32);
      v += __shfl_xor(v, 8, 32);
      lrow[r] = lrow[r] * scl[r] + v;
    }
#pragma unroll
    for (int j = 0; j < 8; ++j)
#pragma unroll
      for (int r = 0; r < 8; ++r) o[j][r] *= scl[r];

    // C-layout P -> bf16 in LDS -> reload in A-fragment layout
#pragma unroll
    for (int j = 0; j < 4; ++j)
#pragma unroll
      for (int r = 0; r < 8; ++r)
        Pb[wid][rb + r][16 * j + colx] = (__bf16)sf[j][r];
    __syncthreads();

#pragma unroll
    for (int kk2 = 0; kk2 < 2; ++kk2) {
      v16bf pa = load_frag_a(&Pb[wid][0][0], 0, kk2 * 32, 64);
#pragma unroll
      for (int j2 = 0; j2 < 8; ++j2) {
        v16bf vb = load_frag_b(v_base, 16 * j2, t0 + kk2 * 32, S_);
        o[j2] = wmma_bf16f32(pa, vb, o[j2]);
      }
    }
    __syncthreads();
  }

#pragma unroll
  for (int j2 = 0; j2 < 8; ++j2)
#pragma unroll
    for (int r = 0; r < 8; ++r) {
      int srow = s0 + rb + r;
      long tok = (long)srow * B_ + b;
      Ctx[tok * H_ + h * HN_ + 16 * j2 + colx] = (__bf16)(o[j2][r] / lrow[r]);
    }
}

// ---------------- host launcher ----------------
extern "C" void kernel_launch(void* const* d_in, const int* in_sizes, int n_in,
                              void* d_out, int out_size, void* d_ws, size_t ws_size,
                              hipStream_t stream) {
  (void)in_sizes; (void)n_in; (void)out_size; (void)ws_size;

  const float* hs      = (const float*)d_in[0];
  // d_in[1] = attention_mask (causal, handled structurally)
  const float* w_qkv   = (const float*)d_in[2];
  const float* b_qkv   = (const float*)d_in[3];
  const float* w_dense = (const float*)d_in[4];
  const float* b_dense = (const float*)d_in[5];
  float* out = (float*)d_out;

  char* ws = (char*)d_ws;
  const size_t szX   = (size_t)NT_ * H_  * 2;   // 16 MiB
  const size_t szWq  = (size_t)H3_ * H_  * 2;   // 24 MiB
  const size_t szWd  = (size_t)H_  * H_  * 2;   //  8 MiB
  const size_t szMix = (size_t)NT_ * H3_ * 2;   // 48 MiB
  const size_t szHead = (size_t)B_ * NP_ * S_ * HN_ * 2;  // 16 MiB

  __bf16* Xbf = (__bf16*)(ws);                              // dead after gemm1
  __bf16* Wqb = (__bf16*)(ws + szX);                        // dead after gemm1
  __bf16* Wdb = (__bf16*)(ws + szX + szWq);
  __bf16* Mix = (__bf16*)(ws + szX + szWq + szWd);
  __bf16* Kh  = (__bf16*)(ws + szX + szWq + szWd + szMix);
  __bf16* Vt  = (__bf16*)(ws + szX + szWq + szWd + szMix + szHead);
  __bf16* Qh  = Wqb;   // reuse Wqkv-bf16 region (gemm1 already consumed it)
  __bf16* Ctx = Xbf;   // reuse Xbf region      (gemm1 already consumed it)
  // total footprint: 128 MiB

  const int nX  = NT_ * H_;
  const int nWq = H3_ * H_;
  const int nWd = H_ * H_;
  cvt_f32_bf16<<<(nX  + 255) / 256, 256, 0, stream>>>(hs,      Xbf, nX);
  cvt_f32_bf16<<<(nWq + 255) / 256, 256, 0, stream>>>(w_qkv,   Wqb, nWq);
  cvt_f32_bf16<<<(nWd + 255) / 256, 256, 0, stream>>>(w_dense, Wdb, nWd);

  // mixed = X @ Wqkv^T + b_qkv   [4096 x 6144]
  gemm_nt<false><<<dim3(H3_ / 128, NT_ / 128), 256, 0, stream>>>(
      Xbf, Wqb, b_qkv, nullptr, Mix, H3_, H_);

  repack_qkv<<<(NT_ * NP_ * HN_ + 255) / 256, 256, 0, stream>>>(Mix, Qh, Kh, Vt);

  flash_attn<<<dim3(S_ / 64, NP_, B_), 128, 0, stream>>>(Qh, Kh, Vt, Ctx);

  // out = Ctx @ Wdense^T + b_dense  [4096 x 2048], f32 to d_out
  gemm_nt<true><<<dim3(H_ / 128, NT_ / 128), 256, 0, stream>>>(
      Ctx, Wdb, b_dense, out, nullptr, H_, H_);
}
